// AVWGCN_3736621548017
// MI455X (gfx1250) — compile-verified, matrix operand order checked by reading, not run on hardware
//
#include <hip/hip_runtime.h>
#include <hip/hip_bf16.h>

// Problem constants (match reference)
#define BATCH   64
#define N_NODES 2048
#define CIN     64
#define COUT    64
#define CHEB    3
#define EMBED   16
#define KI      (CHEB * CIN)   // 192

typedef __attribute__((ext_vector_type(16))) _Float16 v16h;
typedef __attribute__((ext_vector_type(8)))  _Float16 v8h;
typedef __attribute__((ext_vector_type(8)))  float    v8f;

// ---------------------------------------------------------------------------
// Fragment helpers (CDNA5 WMMA 16x16x32 f16 layouts, wave32)
// A 16x32: lane -> row = lane&15, khalf = lane>>4; holds K=[koff..koff+7] and
// [koff+16..koff+23], koff = 8*khalf. Two contiguous 16B loads.
// B 32x16 is the mirrored (column-striped) layout, so we keep the B operand
// transposed in memory and use the identical loader.
// ---------------------------------------------------------------------------
__device__ __forceinline__ v16h load_frag16(const _Float16* __restrict__ p) {
    v8h lo = *reinterpret_cast<const v8h*>(p);
    v8h hi = *reinterpret_cast<const v8h*>(p + 16);
    return __builtin_shufflevector(lo, hi, 0, 1, 2, 3, 4, 5, 6, 7,
                                           8, 9, 10, 11, 12, 13, 14, 15);
}

__device__ __forceinline__ v8f wmma_f16(v16h a, v16h b, v8f c) {
    // 8 args: (neg_a, A, neg_b, B, c_mod, C, reuse_a, reuse_b)
    return __builtin_amdgcn_wmma_f32_16x16x32_f16(false, a, false, b,
                                                  (short)0, c, false, false);
}

// ---------------------------------------------------------------------------
// K1: x (f32, B,N,C) -> x16 (f16, same layout) and xT (f16, B,C,N)
// ---------------------------------------------------------------------------
__global__ void cvt_x_kernel(const float* __restrict__ x,
                             _Float16* __restrict__ x16,
                             _Float16* __restrict__ xT) {
    size_t idx = (size_t)blockIdx.x * blockDim.x + threadIdx.x;
    if (idx >= (size_t)BATCH * N_NODES * CIN) return;
    int c = (int)(idx % CIN);
    int m = (int)((idx / CIN) % N_NODES);
    int b = (int)(idx / ((size_t)CIN * N_NODES));
    _Float16 h = (_Float16)x[idx];
    x16[idx] = h;
    xT[((size_t)b * CIN + c) * N_NODES + m] = h;
}

// ---------------------------------------------------------------------------
// K2: support = softmax(relu(E E^T)) rows; emit f16 S and S^T.
// One 256-thread block per row; row embedding cached, full row staged in LDS.
// ---------------------------------------------------------------------------
__global__ void __launch_bounds__(256)
support_kernel(const float* __restrict__ E,
               _Float16* __restrict__ S,
               _Float16* __restrict__ ST) {
    const int row = blockIdx.x;
    const int t = threadIdx.x;
    __shared__ float er[EMBED];
    __shared__ float red[256];
    __shared__ float vals[N_NODES];

    if (t < EMBED) er[t] = E[row * EMBED + t];
    __syncthreads();

    float lmax = 0.0f;  // relu => values >= 0
    for (int j = t; j < N_NODES; j += 256) {
        const float* ej = E + (size_t)j * EMBED;
        float acc = 0.0f;
#pragma unroll
        for (int d = 0; d < EMBED; ++d) acc += er[d] * ej[d];
        acc = fmaxf(acc, 0.0f);
        vals[j] = acc;
        lmax = fmaxf(lmax, acc);
    }
    red[t] = lmax;
    __syncthreads();
    for (int s = 128; s > 0; s >>= 1) {
        if (t < s) red[t] = fmaxf(red[t], red[t + s]);
        __syncthreads();
    }
    const float mx = red[0];
    __syncthreads();

    float lsum = 0.0f;
    for (int j = t; j < N_NODES; j += 256) {
        float e = __expf(vals[j] - mx);
        vals[j] = e;
        lsum += e;
    }
    red[t] = lsum;
    __syncthreads();
    for (int s = 128; s > 0; s >>= 1) {
        if (t < s) red[t] += red[t + s];
        __syncthreads();
    }
    const float inv = 1.0f / red[0];

    for (int j = t; j < N_NODES; j += 256) {
        _Float16 h = (_Float16)(vals[j] * inv);
        S[(size_t)row * N_NODES + j] = h;
        ST[(size_t)j * N_NODES + row] = h;
    }
}

// ---------------------------------------------------------------------------
// K3: WT[n][o][k*64+i] = sum_d E[n,d] * wp[d,k,i,o]   (f16, N x COUT x 192)
// ---------------------------------------------------------------------------
__global__ void wt_kernel(const float* __restrict__ E,
                          const float* __restrict__ wp,
                          _Float16* __restrict__ WT) {
    size_t tid = (size_t)blockIdx.x * blockDim.x + threadIdx.x;
    if (tid >= (size_t)N_NODES * COUT * KI) return;
    int ki = (int)(tid % KI);
    int o  = (int)((tid / KI) % COUT);
    int n  = (int)(tid / ((size_t)KI * COUT));
    float acc = 0.0f;
#pragma unroll
    for (int d = 0; d < EMBED; ++d)
        acc += E[(size_t)n * EMBED + d] * wp[((size_t)d * KI + ki) * COUT + o];
    WT[tid] = (_Float16)acc;
}

// ---------------------------------------------------------------------------
// K4: T2 = 2 * S @ S - I  (2048x2048x2048 GEMM, f16 in/out, f32 accum)
// One wave per 16x64 output tile: 128 ntiles x 32 col-groups = 4096 waves.
// Software double-buffered K loop: next fragments issued before current WMMAs
// so the per-WMMA wait covers loads a full iteration old.
// ---------------------------------------------------------------------------
__global__ void __launch_bounds__(128)
cheb2_kernel(const _Float16* __restrict__ S,
             const _Float16* __restrict__ ST,
             _Float16* __restrict__ T2) {
    const int lane = threadIdx.x & 31;
    const int gw = blockIdx.x * 4 + (threadIdx.x >> 5);
    const int ntile = gw >> 5;   // 0..127
    const int cg = gw & 31;      // column group of 64
    const int col0 = lane & 15;
    const int koff = (lane >> 4) * 8;

    const _Float16* arow = S + (size_t)(ntile * 16 + col0) * N_NODES + koff;
    const _Float16* brow[4];
#pragma unroll
    for (int j = 0; j < 4; ++j)
        brow[j] = ST + (size_t)(cg * 64 + j * 16 + col0) * N_NODES + koff;

    v8f acc[4] = {v8f{}, v8f{}, v8f{}, v8f{}};

    // prologue: fetch K-step 0
    v16h a_cur = load_frag16(arow);
    v16h b_cur[4];
#pragma unroll
    for (int j = 0; j < 4; ++j) b_cur[j] = load_frag16(brow[j]);

    for (int kk = 0; kk < N_NODES - 32; kk += 32) {
        const int kn = kk + 32;
        // L2 warm-up for the stream, 8 K-steps ahead (global_prefetch_b8)
        __builtin_prefetch(arow + kk + 256, 0, 1);
        // issue next-iteration fragment loads first
        v16h a_nxt = load_frag16(arow + kn);
        v16h b_nxt[4];
#pragma unroll
        for (int j = 0; j < 4; ++j) b_nxt[j] = load_frag16(brow[j] + kn);
        // compute on the fragments loaded last iteration
#pragma unroll
        for (int j = 0; j < 4; ++j) acc[j] = wmma_f16(a_cur, b_cur[j], acc[j]);
        a_cur = a_nxt;
#pragma unroll
        for (int j = 0; j < 4; ++j) b_cur[j] = b_nxt[j];
    }
#pragma unroll
    for (int j = 0; j < 4; ++j) acc[j] = wmma_f16(a_cur, b_cur[j], acc[j]);

    const int mbase = ntile * 16 + (lane >> 4) * 8;
#pragma unroll
    for (int j = 0; j < 4; ++j) {
        const int c = cg * 64 + j * 16 + col0;
#pragma unroll
        for (int r = 0; r < 8; ++r) {
            const int m = mbase + r;
            float v = 2.0f * acc[j][r] - ((m == c) ? 1.0f : 0.0f);
            T2[(size_t)m * N_NODES + c] = (_Float16)v;
        }
    }
}

// ---------------------------------------------------------------------------
// K5: Y[b] = A @ X_b  for all batches (A is S or T2; X_b via xT).
// One wave per (batch, ntile): 64*128 = 8192 waves; wave tile is 16x64 (all C).
// Same double-buffered K loop as K4.
// ---------------------------------------------------------------------------
__global__ void __launch_bounds__(128)
prop_kernel(const _Float16* __restrict__ A,
            const _Float16* __restrict__ xT,
            _Float16* __restrict__ Y) {
    const int lane = threadIdx.x & 31;
    const int gw = blockIdx.x * 4 + (threadIdx.x >> 5);
    const int b = gw >> 7;        // 0..63
    const int ntile = gw & 127;   // 0..127
    const int col0 = lane & 15;
    const int koff = (lane >> 4) * 8;

    const _Float16* xtb = xT + (size_t)b * CIN * N_NODES;
    const _Float16* arow = A + (size_t)(ntile * 16 + col0) * N_NODES + koff;
    const _Float16* brow[4];
#pragma unroll
    for (int j = 0; j < 4; ++j)
        brow[j] = xtb + (size_t)(j * 16 + col0) * N_NODES + koff;

    v8f acc[4] = {v8f{}, v8f{}, v8f{}, v8f{}};

    v16h a_cur = load_frag16(arow);
    v16h b_cur[4];
#pragma unroll
    for (int j = 0; j < 4; ++j) b_cur[j] = load_frag16(brow[j]);

    for (int kk = 0; kk < N_NODES - 32; kk += 32) {
        const int kn = kk + 32;
        __builtin_prefetch(arow + kk + 256, 0, 1);
        v16h a_nxt = load_frag16(arow + kn);
        v16h b_nxt[4];
#pragma unroll
        for (int j = 0; j < 4; ++j) b_nxt[j] = load_frag16(brow[j] + kn);
#pragma unroll
        for (int j = 0; j < 4; ++j) acc[j] = wmma_f16(a_cur, b_cur[j], acc[j]);
        a_cur = a_nxt;
#pragma unroll
        for (int j = 0; j < 4; ++j) b_cur[j] = b_nxt[j];
    }
#pragma unroll
    for (int j = 0; j < 4; ++j) acc[j] = wmma_f16(a_cur, b_cur[j], acc[j]);

    _Float16* yb = Y + ((size_t)b * N_NODES + (size_t)ntile * 16) * CIN;
    const int mbase = (lane >> 4) * 8;
#pragma unroll
    for (int j = 0; j < 4; ++j) {
        const int c = j * 16 + col0;
#pragma unroll
        for (int r = 0; r < 8; ++r)
            yb[(size_t)(mbase + r) * CIN + c] = (_Float16)acc[j][r];
    }
}

// ---------------------------------------------------------------------------
// K6: per-node grouped GEMM: out[b,n,o] = sum_{k,i} feat_k[b,n,i] * W_n[k,i,o]
//      + bias_n[o].   (64 x 192) @ (192 x 64) per node, one block per node,
// 4 waves = 4 output-column tiles, each wave 4 row(batch) tiles.
// ---------------------------------------------------------------------------
__global__ void __launch_bounds__(128)
final_kernel(const _Float16* __restrict__ x16,
             const _Float16* __restrict__ Y1,
             const _Float16* __restrict__ Y2,
             const _Float16* __restrict__ WT,
             const float* __restrict__ E,
             const float* __restrict__ bp,
             float* __restrict__ out) {
    const int n = blockIdx.x;
    const int lane = threadIdx.x & 31;
    const int otile = threadIdx.x >> 5;  // 0..3
    const int col0 = lane & 15;
    const int koff = (lane >> 4) * 8;
    const int o = otile * 16 + col0;

    const _Float16* wrow = WT + ((size_t)n * COUT + o) * KI;
    const _Float16* segs[3] = {x16, Y1, Y2};
    v8f acc[4] = {v8f{}, v8f{}, v8f{}, v8f{}};

#pragma unroll
    for (int kt = 0; kt < 6; ++kt) {               // K = 192 = 6 * 32
        const _Float16* xs = segs[kt >> 1];        // cheb order: I, S, T2
        const int kin = (kt & 1) * 32 + koff;      // offset within 64-wide seg
        v16h bb = load_frag16(wrow + kt * 32 + koff);
#pragma unroll
        for (int t = 0; t < 4; ++t) {
            const int brow = t * 16 + col0;
            const _Float16* ap =
                xs + ((size_t)brow * N_NODES + n) * CIN + kin;
            acc[t] = wmma_f16(load_frag16(ap), bb, acc[t]);
        }
    }

    float bias = 0.0f;
#pragma unroll
    for (int d = 0; d < EMBED; ++d)
        bias += E[(size_t)n * EMBED + d] * bp[(size_t)d * COUT + o];

    const int mbase = (lane >> 4) * 8;
#pragma unroll
    for (int t = 0; t < 4; ++t) {
#pragma unroll
        for (int r = 0; r < 8; ++r) {
            const int b = t * 16 + mbase + r;
            out[((size_t)b * N_NODES + n) * COUT + o] = acc[t][r] + bias;
        }
    }
}

// ---------------------------------------------------------------------------
extern "C" void kernel_launch(void* const* d_in, const int* in_sizes, int n_in,
                              void* d_out, int out_size, void* d_ws,
                              size_t ws_size, hipStream_t stream) {
    (void)in_sizes; (void)n_in; (void)out_size; (void)ws_size;

    const float* x  = (const float*)d_in[0];  // (B, N, CIN)
    const float* E  = (const float*)d_in[1];  // (N, EMBED)
    const float* wp = (const float*)d_in[2];  // (EMBED, CHEB, CIN, COUT)
    const float* bp = (const float*)d_in[3];  // (EMBED, COUT)
    float* out = (float*)d_out;               // (B, N, COUT)

    // Workspace carve-up (f16 staging, ~143 MB total)
    char* ws = (char*)d_ws;
    size_t off = 0;
    auto carve = [&](size_t bytes) {
        char* p = ws + off;
        off += (bytes + 255) & ~(size_t)255;
        return p;
    };
    const size_t nn_h   = (size_t)N_NODES * N_NODES * sizeof(_Float16);
    const size_t bnc_h  = (size_t)BATCH * N_NODES * CIN * sizeof(_Float16);
    _Float16* S16 = (_Float16*)carve(nn_h);
    _Float16* ST16 = (_Float16*)carve(nn_h);
    _Float16* T216 = (_Float16*)carve(nn_h);
    _Float16* x16 = (_Float16*)carve(bnc_h);
    _Float16* xT16 = (_Float16*)carve(bnc_h);
    _Float16* Y1 = (_Float16*)carve(bnc_h);
    _Float16* Y2 = (_Float16*)carve(bnc_h);
    _Float16* WT = (_Float16*)carve((size_t)N_NODES * COUT * KI * sizeof(_Float16));

    // 1) convert + transpose x
    {
        size_t tot = (size_t)BATCH * N_NODES * CIN;
        cvt_x_kernel<<<(unsigned)((tot + 255) / 256), 256, 0, stream>>>(x, x16, xT16);
    }
    // 2) adaptive adjacency: softmax(relu(E E^T)) -> S, S^T
    support_kernel<<<N_NODES, 256, 0, stream>>>(E, S16, ST16);
    // 3) per-node weights (transposed for WMMA B operand)
    {
        size_t tot = (size_t)N_NODES * COUT * KI;
        wt_kernel<<<(unsigned)((tot + 255) / 256), 256, 0, stream>>>(E, wp, WT);
    }
    // 4) Chebyshev T2 = 2 S S - I
    cheb2_kernel<<<(128 * 32) / 4, 128, 0, stream>>>(S16, ST16, T216);
    // 5) feature propagation: Y1 = S x, Y2 = T2 x (per batch)
    prop_kernel<<<(BATCH * 128) / 4, 128, 0, stream>>>(S16, xT16, Y1);
    prop_kernel<<<(BATCH * 128) / 4, 128, 0, stream>>>(T216, xT16, Y2);
    // 6) per-node grouped matmul + bias -> fp32 output
    final_kernel<<<N_NODES, 128, 0, stream>>>(x16, Y1, Y2, WT, E, bp, out);
}
